// LightningAttention_80032420593966
// MI455X (gfx1250) — matrix operand order repaired and checked by reading, not compiled, not run on hardware
//
#include <hip/hip_runtime.h>

// ---------------------------------------------------------------------------
// LightningAttention (non-causal linear attention) for MI455X / gfx1250.
// B=8, T=8192, D=128, H=8, Dh=16.
// bf16 WMMA (v_wmma_f32_16x16x32_bf16) for all four GEMMs and the per-head
// 16x16 context reduction (time = K dimension). Weight tiles are DMA'd into
// LDS with global_load_async_to_lds_b128 (ASYNCcnt). Two passes over the
// [B,T,D] tensors -> bandwidth-limited (~130 MB @ 23.3 TB/s).
// ---------------------------------------------------------------------------

typedef __attribute__((ext_vector_type(8)))  float  v8f;
typedef __attribute__((ext_vector_type(16))) __bf16 v16bf;

union ABFrag { v16bf v; unsigned int u[8]; };

constexpr int Bsz = 8;
constexpr int T   = 8192;
constexpr int D   = 128;
constexpr int H   = 8;
constexpr int TT  = 128;       // rows (timesteps) per workgroup tile
constexpr int TP  = TT + 8;    // padded pitch for transposed k/v (bank spread)
constexpr int KP  = D / 2;     // bf16 pairs per row
constexpr int NTHR = 256;      // 8 waves (wave32)
constexpr int NWAVES = 8;

__device__ __forceinline__ unsigned short f2bf(float f) {
  unsigned int u = __builtin_bit_cast(unsigned int, f);
  u += 0x7FFFu + ((u >> 16) & 1u);            // round-to-nearest-even
  return (unsigned short)(u >> 16);
}
__device__ __forceinline__ float bf2f(unsigned short s) {
  return __builtin_bit_cast(float, ((unsigned int)s) << 16);
}

// Async DMA copy of a bf16 weight tile (global -> LDS), 16B per lane-op.
// GVS mode: addr = SGPR64 + VGPR32 + inst_offset. VDST = LDS byte address.
__device__ __forceinline__ void async_copy_16(void* dst, const void* src,
                                              int bytes, int tid) {
  for (int i = tid; i < bytes / 16; i += NTHR) {
    unsigned int lds = (unsigned int)(unsigned long long)
                       ((char*)dst + (size_t)i * 16);
    asm volatile("global_load_async_to_lds_b128 %0, %1, %2"
                 :: "v"(lds), "v"(i * 16), "s"(src) : "memory");
  }
  asm volatile("s_wait_asynccnt 0x0" ::: "memory");
}

// ---------------------------------------------------------------------------
// Kernel 0: zero accumulators; convert W matrices to bf16, transposed
// (wt[m][n][k] = W_m[k][n]) so WMMA B-fragments are contiguous-pair LDS loads.
// ---------------------------------------------------------------------------
__global__ __launch_bounds__(256) void prep_kernel(
    const float* __restrict__ Wq, const float* __restrict__ Wk,
    const float* __restrict__ Wv, const float* __restrict__ Wo,
    unsigned short* __restrict__ wt, float* __restrict__ ctx,
    float* __restrict__ ksum) {
  unsigned int i = blockIdx.x * blockDim.x + threadIdx.x;
  if (i < Bsz * H * 16 * 16) ctx[i] = 0.0f;
  if (i < Bsz * H * 16)      ksum[i] = 0.0f;
  if (i < 4u * D * D) {
    int m = i / (D * D);
    int r = i % (D * D);
    int n = r / D;          // output column (row of W^T)
    int k = r % D;          // input index
    const float* Ws = (m == 0) ? Wq : (m == 1) ? Wk : (m == 2) ? Wv : Wo;
    wt[i] = f2bf(Ws[k * D + n]);
  }
}

// ---------------------------------------------------------------------------
// One QKV GEMM pass (M = 0:q, 1:k, 2:v), fully specialized: WMMA + bias +
// (RoPE + elu+1 for q/k). q -> global f32; k/v -> LDS bf16 *transposed*
// (kbT[col][t]) so the context WMMA gets contiguous-pair operands.
// ---------------------------------------------------------------------------
template <int M>
__device__ __forceinline__ void qkv_pass(
    const unsigned int (*xb)[KP], const unsigned int (*wb)[KP],
    unsigned short* kbT, unsigned short* vbT, const float* bias,
    float* qout, int b, int row0, int lane, int wave) {
  const int h8 = lane >> 4;
  const int nl = lane & 15;
  for (int tix = wave; tix < 64; tix += NWAVES) {
    const int rt = tix >> 3, ct = tix & 7;
    v8f acc = {};
    const int mrow = rt * 16 + nl;
    for (int kt = 0; kt < 4; ++kt) {
      ABFrag a, bm;
#pragma unroll
      for (int v = 0; v < 8; ++v) {
        // A 16x32 bf16: lane<16 -> K 0-7 & 16-23; lane>=16 -> 8-15 & 24-31
        a.u[v]  = xb[mrow][kt * 16 + ((v >= 4) ? 8 : 0) + h8 * 4 + (v & 3)];
        // B 32x16 bf16: lane<16 -> K 0-7 & 16-23; lane>=16 -> 8-15 & 24-31
        bm.u[v] = wb[ct * 16 + nl][kt * 16 + ((v >= 4) ? 8 : 0) + h8 * 4 + (v & 3)];
      }
      acc = __builtin_amdgcn_wmma_f32_16x16x32_bf16(
          false, a.v, false, bm.v, (short)0, acc, false, false);
    }
    const int col = ct * 16 + nl;          // each 16-col tile == one head
    const float bval = bias[col];
    const bool  lo = (nl < 8);
    const int   fi = nl & 7;
    // inv_freq = 10000^(-fi/8) = exp(-fi * ln(10000)/8)
    const float invf = __expf(-(float)fi * (9.210340371976184f * 0.125f));
#pragma unroll
    for (int r = 0; r < 8; ++r) {
      float val = acc[r] + bval;
      const int tl = rt * 16 + r + (h8 ? 8 : 0);
      if constexpr (M == 2) {              // v: bias only, transposed store
        vbT[col * TP + tl] = f2bf(val);
      } else {
        float par = __shfl_xor(val, 8, 32);  // paired rotary component
        float sn, cs;
        __sincosf((float)(row0 + tl) * invf, &sn, &cs);
        float rv = val * cs + (lo ? -par * sn : par * sn);
        rv = (rv > 0.0f) ? (rv + 1.0f) : __expf(rv);  // elu(x)+1
        if constexpr (M == 0)
          qout[((size_t)b * T + row0 + tl) * D + col] = rv;
        else
          kbT[col * TP + tl] = f2bf(rv);
      }
    }
  }
}

// ---------------------------------------------------------------------------
// Kernel 1: QKV projection + activations; then per-head context = K^T V via
// WMMA with time as K dimension; atomic f32 reduce to global.
// ---------------------------------------------------------------------------
__global__ __launch_bounds__(NTHR) void qkv_ctx_kernel(
    const float* __restrict__ x,
    const float* __restrict__ bq, const float* __restrict__ bk,
    const float* __restrict__ bv,
    const unsigned short* __restrict__ wt,
    float* __restrict__ qout, float* __restrict__ ctx,
    float* __restrict__ ksum) {
  __shared__ unsigned int xb[TT][KP];                       // 32 KB
  __shared__ unsigned int wb[D][KP];                        // 32 KB
  __shared__ __align__(16) unsigned short kbT[D * TP];      // ~34 KB (transposed)
  __shared__ __align__(16) unsigned short vbT[D * TP];      // ~34 KB (transposed)

  const int b    = blockIdx.y;
  const int tile = blockIdx.x;
  const int row0 = tile * TT;
  const int tid  = threadIdx.x;
  const int lane = tid & 31;
  const int wave = tid >> 5;

  // ---- load x tile -> bf16 pairs in LDS; prefetch next tile ----
  for (int i = tid; i < TT * KP; i += NTHR) {
    int r = i / KP, p = i % KP;
    const float* src = x + (((size_t)b * T + row0 + r) * D + p * 2);
    float2 f = *(const float2*)src;
    xb[r][p] = (unsigned int)f2bf(f.x) | ((unsigned int)f2bf(f.y) << 16);
    if (tile + 1 < (int)gridDim.x)
      __builtin_prefetch(src + (size_t)TT * D, 0, 0);  // global_prefetch_b8
  }

#pragma unroll
  for (int m = 0; m < 3; ++m) {
    __syncthreads();                      // wb reuse + (iter0) xb visibility
    async_copy_16(&wb[0][0], wt + (size_t)m * D * D, D * D * 2, tid);
    __syncthreads();
    if (m == 0)
      qkv_pass<0>(xb, wb, kbT, vbT, bq, qout, b, row0, lane, wave);
    else if (m == 1)
      qkv_pass<1>(xb, wb, kbT, vbT, bk, qout, b, row0, lane, wave);
    else
      qkv_pass<2>(xb, wb, kbT, vbT, bv, qout, b, row0, lane, wave);
  }
  __syncthreads();

  // ---- context: head h (= wave): ctx[d][e] += sum_t k[t,d] v[t,e]
  //      WMMA, time as K-dim: A = K^T (16x32), B = V (32x16); both operands
  //      are contiguous-pair reads from the transposed LDS tiles.
  {
    const int h  = wave;
    const int h8 = lane >> 4;
    const int nl = lane & 15;
    v8f cacc = {};
    const unsigned int* krow =
        (const unsigned int*)(kbT + (h * 16 + nl) * TP);
    const unsigned int* vrow =
        (const unsigned int*)(vbT + (h * 16 + nl) * TP);
    for (int kt = 0; kt < TT / 32; ++kt) {
      ABFrag a, bm;
#pragma unroll
      for (int v = 0; v < 8; ++v) {
        a.u[v]  = krow[kt * 16 + ((v >= 4) ? 8 : 0) + h8 * 4 + (v & 3)];
        bm.u[v] = vrow[kt * 16 + ((v >= 4) ? 8 : 0) + h8 * 4 + (v & 3)];
      }
      cacc = __builtin_amdgcn_wmma_f32_16x16x32_bf16(
          false, a.v, false, bm.v, (short)0, cacc, false, false);
    }
    float* cb = ctx + ((size_t)b * H + h) * 256;
#pragma unroll
    for (int r = 0; r < 8; ++r)
      atomicAdd(&cb[(r + 8 * h8) * 16 + nl], cacc[r]);

    // k_sum over this tile's timesteps (lanes 0..15 carry d; contiguous row)
    if (lane < 16) {
      const unsigned short* kr = kbT + (h * 16 + lane) * TP;
      float s = 0.0f;
      for (int t = 0; t < TT; ++t) s += bf2f(kr[t]);
      atomicAdd(ksum + ((size_t)b * H + h) * 16 + lane, s);
    }
  }
}

// ---------------------------------------------------------------------------
// Kernel 2: out = ((q @ ctx) * 1/(q.ksum + 1e-6)) @ Wo + bo
// attention core in f32 VALU (tiny), final GEMM via WMMA from LDS bf16.
// ---------------------------------------------------------------------------
__global__ __launch_bounds__(NTHR) void attn_out_kernel(
    const float* __restrict__ qout, const float* __restrict__ ctx,
    const float* __restrict__ ksum, const unsigned short* __restrict__ wt,
    const float* __restrict__ bo, float* __restrict__ out) {
  __shared__ float        ctx_s[H][16][16];   // 8 KB
  __shared__ float        ks_s[H][16];        // 0.5 KB
  __shared__ unsigned int wob[D][KP];         // Wo^T bf16 (32 KB)
  __shared__ unsigned int ab[TT][KP];         // attn tile bf16 (32 KB)

  const int b    = blockIdx.y;
  const int tile = blockIdx.x;
  const int row0 = tile * TT;
  const int tid  = threadIdx.x;
  const int lane = tid & 31;
  const int wave = tid >> 5;

  async_copy_16(&wob[0][0], wt + (size_t)3 * D * D, D * D * 2, tid);
  for (int i = tid; i < H * 256; i += NTHR)
    (&ctx_s[0][0][0])[i] = ctx[(size_t)b * H * 256 + i];
  if (tid < H * 16) (&ks_s[0][0])[tid] = ksum[(size_t)b * H * 16 + tid];
  __syncthreads();

  // attention core: one (t, h) pair per iteration
  for (int p = tid; p < TT * H; p += NTHR) {
    const int tl = p >> 3, h = p & 7;
    const float* qr = qout + ((size_t)b * T + row0 + tl) * D + h * 16;
    float qv[16];
#pragma unroll
    for (int d = 0; d < 16; d += 4) {
      float4 f = *(const float4*)(qr + d);
      qv[d] = f.x; qv[d + 1] = f.y; qv[d + 2] = f.z; qv[d + 3] = f.w;
    }
    float zden = 1e-6f;
#pragma unroll
    for (int d = 0; d < 16; ++d) zden += qv[d] * ks_s[h][d];
    const float z = 1.0f / zden;
    unsigned short* arow = (unsigned short*)&ab[tl][0];
#pragma unroll
    for (int e = 0; e < 16; ++e) {
      float s = 0.0f;
#pragma unroll
      for (int d = 0; d < 16; ++d) s += qv[d] * ctx_s[h][d][e];
      arow[h * 16 + e] = f2bf(s * z);
    }
  }
  __syncthreads();

  // final projection: attn[128x128] @ Wo + bo
  const int h8 = lane >> 4;
  const int nl = lane & 15;
  for (int tix = wave; tix < 64; tix += NWAVES) {
    const int rt = tix >> 3, ct = tix & 7;
    v8f acc = {};
    const int mrow = rt * 16 + nl;
    for (int kt = 0; kt < 4; ++kt) {
      ABFrag a, bm;
#pragma unroll
      for (int v = 0; v < 8; ++v) {
        a.u[v]  = ab[mrow][kt * 16 + ((v >= 4) ? 8 : 0) + h8 * 4 + (v & 3)];
        bm.u[v] = wob[ct * 16 + nl][kt * 16 + ((v >= 4) ? 8 : 0) + h8 * 4 + (v & 3)];
      }
      acc = __builtin_amdgcn_wmma_f32_16x16x32_bf16(
          false, a.v, false, bm.v, (short)0, acc, false, false);
    }
    const int col = ct * 16 + nl;
    const float bb = bo[col];
#pragma unroll
    for (int r = 0; r < 8; ++r)
      out[((size_t)b * T + row0 + rt * 16 + r + 8 * h8) * D + col] =
          acc[r] + bb;
  }
}

// ---------------------------------------------------------------------------
extern "C" void kernel_launch(void* const* d_in, const int* in_sizes, int n_in,
                              void* d_out, int out_size, void* d_ws,
                              size_t ws_size, hipStream_t stream) {
  (void)in_sizes; (void)n_in; (void)out_size; (void)ws_size;
  const float* x  = (const float*)d_in[0];
  const float* Wq = (const float*)d_in[1];
  const float* bq = (const float*)d_in[2];
  const float* Wk = (const float*)d_in[3];
  const float* bk = (const float*)d_in[4];
  const float* Wv = (const float*)d_in[5];
  const float* bv = (const float*)d_in[6];
  const float* Wo = (const float*)d_in[7];
  const float* bo = (const float*)d_in[8];
  float* out = (float*)d_out;

  char* ws = (char*)d_ws;
  float* qbuf = (float*)ws;                               // B*T*D f32
  size_t off = (size_t)Bsz * T * D * sizeof(float);
  float* ctx = (float*)(ws + off);                        // B*H*16*16 f32
  off += (size_t)Bsz * H * 256 * sizeof(float);
  float* ksm = (float*)(ws + off);                        // B*H*16 f32
  off += (size_t)Bsz * H * 16 * sizeof(float);
  unsigned short* wt = (unsigned short*)(ws + off);       // 4*D*D bf16

  prep_kernel<<<(4 * D * D + 255) / 256, 256, 0, stream>>>(
      Wq, Wk, Wv, Wo, wt, ctx, ksm);

  dim3 grid(T / TT, Bsz);
  qkv_ctx_kernel<<<grid, NTHR, 0, stream>>>(x, bq, bk, bv, wt, qbuf, ctx, ksm);
  attn_out_kernel<<<grid, NTHR, 0, stream>>>(qbuf, ctx, ksm, wt, bo, out);
}
